// get_model_154618823035
// MI455X (gfx1250) — matrix operand order repaired and checked by reading, not compiled, not run on hardware
//
#include <hip/hip_runtime.h>
#include <hip/hip_bf16.h>

// ---------------- model constants ----------------
#define BATCH 16
#define NPTS  4096
#define NCLS  16
#define EPS_BN 1e-5f

typedef __attribute__((ext_vector_type(16))) _Float16 v16h;
typedef __attribute__((ext_vector_type(8)))  _Float16 v8h;
typedef __attribute__((ext_vector_type(8)))  float    v8f;

// MLP chain params: weights pre-packed into WMMA B-fragment order (f16),
// epilogue pre-folded to y = relu(acc*es + eb), zero-padded to tile grid.
struct Mlp {
  int nlayers;
  int cin;                 // unpadded input channels
  int cout[3];
  const _Float16* pW[3];   // packed fragments: ((nt*nK+kc)*32+lane)*16+j
  const float* es[3];      // scale, padded to pad32(cout)
  const float* eb[3];      // shift, padded to pad32(cout)
};

// =====================================================================
// Weight pre-pack: f32 (cout,cin) row-major -> f16 B-fragment order.
// n = nt*16 + (lane&15);  kg = kc*32 + 8*(lane>>4) + (j&7) + 16*(j>>3)
// =====================================================================
__global__ void pack_w_kernel(const float* __restrict__ W, _Float16* __restrict__ pW,
                              int cin, int cout)
{
  const int nK = (cin + 31) >> 5;
  const int nT = ((cout + 31) & ~31) >> 4;
  const int total = nT * nK * 512;
  int idx = blockIdx.x * 256 + threadIdx.x;
  if (idx >= total) return;
  const int j    = idx & 15;
  const int lane = (idx >> 4) & 31;
  const int rest = idx >> 9;
  const int kc   = rest % nK;
  const int nt   = rest / nK;
  const int n  = nt * 16 + (lane & 15);
  const int kg = kc * 32 + 8 * (lane >> 4) + (j & 7) + ((j >> 3) << 4);
  float v = (n < cout && kg < cin) ? W[(size_t)n * cin + kg] : 0.f;
  pW[idx] = (_Float16)v;
}

// =====================================================================
// Epilogue pre-fold: es = gamma*rsqrt(var+eps); eb = (bias-mean)*es + beta
// =====================================================================
__global__ void pack_ep_kernel(const float* __restrict__ bias, const float* __restrict__ gamma,
                               const float* __restrict__ beta, const float* __restrict__ mean,
                               const float* __restrict__ var,
                               float* __restrict__ es, float* __restrict__ eb,
                               int cout, int coutPad)
{
  int c = blockIdx.x * 256 + threadIdx.x;
  if (c >= coutPad) return;
  if (c < cout) {
    float s = gamma[c] * rsqrtf(var[c] + EPS_BN);
    es[c] = s;
    eb[c] = (bias[c] - mean[c]) * s + beta[c];
  } else {
    es[c] = 0.f;
    eb[c] = 0.f;
  }
}

// =====================================================================
// One MLP layer on LDS-resident features via V_WMMA_F32_16X16X32_F16.
// Branch-free hot loop: 2x ds_load_b128 (A) + 2x global_load_b128 (B) + wmma.
// =====================================================================
__device__ __forceinline__ void mlp_layer_wmma(
    const _Float16* featIn, _Float16* featOut,
    int nrows, int stride, int cin, int cout,
    const _Float16* pW, const float* es, const float* eb,
    int lane, int wave, int nwaves)
{
  const int nK = (cin + 31) >> 5;            // K chunks of 32
  const int nT = ((cout + 31) & ~31) >> 4;   // N tiles incl. zero pad
  const int nM = nrows >> 4;
  const int lo = lane & 15;
  const int hi = lane >> 4;

  for (int mt = 0; mt < nM; ++mt) {
    const _Float16* arow = featIn + (size_t)(mt * 16 + lo) * stride + 8 * hi;
    for (int nt = wave; nt < nT; nt += nwaves) {
      v8f acc = {};
      const _Float16* wptr = pW + ((size_t)nt * nK) * 512 + lane * 16;
      for (int kc = 0; kc < nK; ++kc) {
        // ---- A fragment: halfs [0..7] and [16..23] from row (16B aligned)
        const v8h* ap = (const v8h*)(arow + (kc << 5));
        v16h a = __builtin_shufflevector(ap[0], ap[2],
                                         0, 1, 2, 3, 4, 5, 6, 7,
                                         8, 9, 10, 11, 12, 13, 14, 15);
        // ---- B fragment: 16 contiguous pre-packed halfs (16B aligned)
        const v8h* wp = (const v8h*)(wptr + ((size_t)kc << 9));
        v16h bf = __builtin_shufflevector(wp[0], wp[1],
                                          0, 1, 2, 3, 4, 5, 6, 7,
                                          8, 9, 10, 11, 12, 13, 14, 15);
        acc = __builtin_amdgcn_wmma_f32_16x16x32_f16(
            false, a, false, bf, (short)0, acc, false, false);
      }
      // ---- folded BN epilogue (padded arrays -> branch-free)
      const int c = nt * 16 + lo;
      const float s = es[c];
      const float b = eb[c];
      #pragma unroll
      for (int r = 0; r < 8; ++r) {
        const int m = mt * 16 + r + 8 * hi;
        featOut[(size_t)m * stride + c] = (_Float16)fmaxf(acc[r] * s + b, 0.f);
      }
    }
  }
}

// =====================================================================
// Transpose input (B,9,N) -> l0_pts (B,N,9) and l0_xyz (B,N,3)
// =====================================================================
__global__ void transpose_in_kernel(const float* __restrict__ xin,
                                    float* __restrict__ l0_xyz,
                                    float* __restrict__ l0_pts)
{
  int g = blockIdx.x * blockDim.x + threadIdx.x;
  if (g >= BATCH * NPTS) return;
  int b = g / NPTS, i = g % NPTS;
  const float* src = xin + (size_t)b * 9 * NPTS + i;
  #pragma unroll
  for (int c = 0; c < 9; ++c) l0_pts[(size_t)g * 9 + c] = src[(size_t)c * NPTS];
  #pragma unroll
  for (int c = 0; c < 3; ++c) l0_xyz[(size_t)g * 3 + c] = src[(size_t)c * NPTS];
}

// =====================================================================
// Farthest point sampling: one block per batch, serial over npoint.
// =====================================================================
__global__ void fps_kernel(const float* __restrict__ xyz, int* __restrict__ fidx,
                           float* __restrict__ dist, int n, int np)
{
  const int b = blockIdx.x, tid = threadIdx.x;
  __shared__ float sv[256];
  __shared__ int   si[256];
  __shared__ int   cur;
  const float* X = xyz + (size_t)b * n * 3;
  float* D = dist + (size_t)b * n;
  for (int i = tid; i < n; i += 256) D[i] = 1e10f;
  if (tid == 0) cur = 0;
  __syncthreads();
  for (int it = 0; it < np; ++it) {
    const int f = cur;
    if (tid == 0) fidx[b * np + it] = f;
    const float cx = X[f * 3 + 0], cy = X[f * 3 + 1], cz = X[f * 3 + 2];
    float bv = -1.f; int bi = 0;
    for (int i = tid; i < n; i += 256) {
      float dx = X[i * 3 + 0] - cx, dy = X[i * 3 + 1] - cy, dz = X[i * 3 + 2] - cz;
      float d = fminf(D[i], dx * dx + dy * dy + dz * dz);
      D[i] = d;
      if (d > bv) { bv = d; bi = i; }
    }
    sv[tid] = bv; si[tid] = bi;
    __syncthreads();
    for (int s = 128; s > 0; s >>= 1) {
      if (tid < s && sv[tid + s] > sv[tid]) { sv[tid] = sv[tid + s]; si[tid] = si[tid + s]; }
      __syncthreads();
    }
    if (tid == 0) cur = si[0];
    __syncthreads();
  }
}

__global__ void gather3_kernel(const float* __restrict__ xyz, const int* __restrict__ fidx,
                               float* __restrict__ out, int n, int np)
{
  int g = blockIdx.x * blockDim.x + threadIdx.x;
  if (g >= BATCH * np) return;
  int b = g / np;
  int src = fidx[g];
  const float* p = xyz + ((size_t)b * n + src) * 3;
  out[(size_t)g * 3 + 0] = p[0];
  out[(size_t)g * 3 + 1] = p[1];
  out[(size_t)g * 3 + 2] = p[2];
}

// =====================================================================
// Ball query: first k indices (ascending) within radius; pad w/ first.
// =====================================================================
__global__ void ball_query_kernel(const float* __restrict__ xyz,
                                  const float* __restrict__ new_xyz,
                                  int* __restrict__ gidx,
                                  int n, int np, int k, float r2)
{
  int g = blockIdx.x * blockDim.x + threadIdx.x;
  if (g >= BATCH * np) return;
  int b = g / np;
  const float* q = new_xyz + (size_t)g * 3;
  const float qx = q[0], qy = q[1], qz = q[2];
  const float* X = xyz + (size_t)b * n * 3;
  int* out = gidx + (size_t)g * k;
  int cnt = 0;
  for (int j = 0; j < n && cnt < k; ++j) {
    float dx = X[j * 3 + 0] - qx, dy = X[j * 3 + 1] - qy, dz = X[j * 3 + 2] - qz;
    if (dx * dx + dy * dy + dz * dz <= r2) out[cnt++] = j;
  }
  const int first = (cnt > 0) ? out[0] : 0;
  for (; cnt < k; ++cnt) out[cnt] = first;
}

// =====================================================================
// Fused SA branch: gather group -> 3-layer WMMA MLP in LDS -> maxpool.
// One block (128 thr = 4 waves) per (batch, center).
// =====================================================================
__global__ void sa_branch_kernel(const float* __restrict__ xyz,
                                 const float* __restrict__ points,
                                 const float* __restrict__ new_xyz,
                                 const int* __restrict__ gidx,
                                 Mlp mp, float* __restrict__ out,
                                 int n, int np, int k, int cin_pts, int stride,
                                 int out_stride, int out_off)
{
  extern __shared__ unsigned char s_raw[];
  _Float16* bufA = (_Float16*)s_raw;
  _Float16* bufB = bufA + (size_t)k * stride;

  const int g = blockIdx.x;
  const int b = g / np;
  const int tid = threadIdx.x, lane = tid & 31, wave = tid >> 5, nwaves = blockDim.x >> 5;
  const int cin0 = cin_pts + 3;
  const int* gi = gidx + (size_t)g * k;
  const float* nx = new_xyz + (size_t)g * 3;
  const float cx = nx[0], cy = nx[1], cz = nx[2];

  // gather + relative xyz + zero pad into LDS (f16)
  for (int s = wave; s < k; s += nwaves) {
    const int src = gi[s];
    const float* ps = points ? points + ((size_t)b * n + src) * cin_pts : nullptr;
    const float* xs = xyz + ((size_t)b * n + src) * 3;
    for (int c = lane; c < stride; c += 32) {
      float v = 0.f;
      if (c < cin_pts) v = ps[c];
      else if (c < cin0) {
        const int d = c - cin_pts;
        const float ctr = (d == 0) ? cx : ((d == 1) ? cy : cz);
        v = xs[d] - ctr;
      }
      bufA[(size_t)s * stride + c] = (_Float16)v;
    }
  }
  __syncthreads();

  _Float16* fin = bufA;
  _Float16* fout = bufB;
  int cin = cin0;
  for (int L = 0; L < mp.nlayers; ++L) {
    mlp_layer_wmma(fin, fout, k, stride, cin, mp.cout[L],
                   mp.pW[L], mp.es[L], mp.eb[L], lane, wave, nwaves);
    __syncthreads();
    _Float16* t = fin; fin = fout; fout = t;
    cin = mp.cout[L];
  }

  // maxpool over the k samples, write pooled feature
  for (int c = tid; c < cin; c += blockDim.x) {
    float m = 0.f;  // post-ReLU >= 0
    for (int s = 0; s < k; ++s) m = fmaxf(m, (float)fin[(size_t)s * stride + c]);
    out[(size_t)g * out_stride + out_off + c] = m;
  }
}

// =====================================================================
// 3-NN inverse-distance interpolation (feature propagation step 1).
// One wave per destination point.
// =====================================================================
__global__ void interp3_kernel(const float* __restrict__ xyz1,
                               const float* __restrict__ xyz2,
                               const float* __restrict__ pts2,
                               float* __restrict__ out, int n1, int s, int c2)
{
  const int g = blockIdx.x;
  const int b = g / n1;
  __shared__ int   idx3[3];
  __shared__ float w3[3];
  if (threadIdx.x == 0) {
    const float* p = xyz1 + (size_t)g * 3;
    const float px = p[0], py = p[1], pz = p[2];
    const float* q = xyz2 + (size_t)b * s * 3;
    float d0 = 1e30f, d1 = 1e30f, d2 = 1e30f;
    int i0 = 0, i1 = 0, i2 = 0;
    for (int j = 0; j < s; ++j) {
      float dx = px - q[j * 3 + 0], dy = py - q[j * 3 + 1], dz = pz - q[j * 3 + 2];
      float d = dx * dx + dy * dy + dz * dz;
      if (d < d0)      { d2 = d1; i2 = i1; d1 = d0; i1 = i0; d0 = d; i0 = j; }
      else if (d < d1) { d2 = d1; i2 = i1; d1 = d; i1 = j; }
      else if (d < d2) { d2 = d; i2 = j; }
    }
    float w0 = 1.f / (d0 + 1e-8f), w1 = 1.f / (d1 + 1e-8f), w2 = 1.f / (d2 + 1e-8f);
    float ws = w0 + w1 + w2;
    idx3[0] = i0; idx3[1] = i1; idx3[2] = i2;
    w3[0] = w0 / ws; w3[1] = w1 / ws; w3[2] = w2 / ws;
  }
  __syncthreads();
  const float* f0 = pts2 + ((size_t)b * s + idx3[0]) * c2;
  const float* f1 = pts2 + ((size_t)b * s + idx3[1]) * c2;
  const float* f2 = pts2 + ((size_t)b * s + idx3[2]) * c2;
  const float a0 = w3[0], a1 = w3[1], a2 = w3[2];
  for (int c = threadIdx.x; c < c2; c += 32)
    out[(size_t)g * c2 + c] = a0 * f0[c] + a1 * f1[c] + a2 * f2[c];
}

// =====================================================================
// Point-wise fused WMMA MLP over 16-point tiles (FP stages + head).
// =====================================================================
__global__ void fp_mlp_kernel(const float* __restrict__ pts1, int c1,
                              const float* __restrict__ pts2i, int c2,
                              Mlp mp, float* __restrict__ out, int stride)
{
  extern __shared__ unsigned char s_raw[];
  _Float16* bufA = (_Float16*)s_raw;
  _Float16* bufB = bufA + (size_t)16 * stride;

  const int prow = blockIdx.x * 16;
  const int tid = threadIdx.x, lane = tid & 31, wave = tid >> 5, nwaves = blockDim.x >> 5;
  const int cin0 = c1 + c2;

  for (int s = wave; s < 16; s += nwaves) {
    const size_t p = (size_t)(prow + s);
    for (int c = lane; c < stride; c += 32) {
      float v = 0.f;
      if (c < c1) v = pts1[p * c1 + c];
      else if (c < cin0) v = pts2i[p * c2 + (c - c1)];
      bufA[(size_t)s * stride + c] = (_Float16)v;
    }
  }
  __syncthreads();

  _Float16* fin = bufA;
  _Float16* fout = bufB;
  int cin = cin0;
  for (int L = 0; L < mp.nlayers; ++L) {
    mlp_layer_wmma(fin, fout, 16, stride, cin, mp.cout[L],
                   mp.pW[L], mp.es[L], mp.eb[L], lane, wave, nwaves);
    __syncthreads();
    _Float16* t = fin; fin = fout; fout = t;
    cin = mp.cout[L];
  }

  const int coutL = cin;
  for (int c = tid; c < 16 * coutL; c += blockDim.x) {
    const int s = c / coutL, ch = c % coutL;
    out[((size_t)prow + s) * coutL + ch] = (float)fin[(size_t)s * stride + ch];
  }
}

// =====================================================================
// Final 128 -> 16 linear (conv2)
// =====================================================================
__global__ void conv2_kernel(const float* __restrict__ h,
                             const float* __restrict__ W2,
                             const float* __restrict__ b2,
                             float* __restrict__ logits)
{
  int g = blockIdx.x * blockDim.x + threadIdx.x;
  if (g >= BATCH * NPTS) return;
  const float* hp = h + (size_t)g * 128;
  #pragma unroll 1
  for (int c = 0; c < NCLS; ++c) {
    const float* wr = W2 + (size_t)c * 128;
    float s = b2[c];
    #pragma unroll 8
    for (int j = 0; j < 128; ++j) s += hp[j] * wr[j];
    logits[(size_t)g * NCLS + c] = s;
  }
}

// =====================================================================
// Softmax over point axis + keypoint einsum + transposed output write.
// One block per (batch, class).
// =====================================================================
__global__ void softmax_kp_kernel(const float* __restrict__ logits,
                                  const float* __restrict__ xyz0,
                                  float* __restrict__ xT, float* __restrict__ kp, int n)
{
  const int b = blockIdx.x / NCLS, c = blockIdx.x % NCLS;
  const int tid = threadIdx.x;
  __shared__ float red[256];
  __shared__ float sM, sS;
  const float* L = logits + (size_t)b * n * NCLS + c;

  float m = -1e30f;
  for (int i = tid; i < n; i += 256) m = fmaxf(m, L[(size_t)i * NCLS]);
  red[tid] = m; __syncthreads();
  for (int s = 128; s > 0; s >>= 1) {
    if (tid < s) red[tid] = fmaxf(red[tid], red[tid + s]);
    __syncthreads();
  }
  if (tid == 0) sM = red[0];
  __syncthreads();
  const float M = sM;

  float acc = 0.f;
  for (int i = tid; i < n; i += 256) acc += __expf(L[(size_t)i * NCLS] - M);
  red[tid] = acc; __syncthreads();
  for (int s = 128; s > 0; s >>= 1) {
    if (tid < s) red[tid] += red[tid + s];
    __syncthreads();
  }
  if (tid == 0) sS = red[0];
  __syncthreads();
  const float inv = 1.f / sS;

  float k0 = 0.f, k1 = 0.f, k2 = 0.f;
  float* xrow = xT + ((size_t)b * NCLS + c) * n;
  const float* X = xyz0 + (size_t)b * n * 3;
  for (int i = tid; i < n; i += 256) {
    const float x = __expf(L[(size_t)i * NCLS] - M) * inv;
    xrow[i] = x;
    k0 += x * X[i * 3 + 0];
    k1 += x * X[i * 3 + 1];
    k2 += x * X[i * 3 + 2];
  }
  float vals[3] = {k0, k1, k2};
  for (int d = 0; d < 3; ++d) {
    red[tid] = vals[d]; __syncthreads();
    for (int s = 128; s > 0; s >>= 1) {
      if (tid < s) red[tid] += red[tid + s];
      __syncthreads();
    }
    if (tid == 0) kp[((size_t)b * NCLS + c) * 3 + d] = red[0];
    __syncthreads();
  }
}

// =====================================================================
// Host driver
// =====================================================================
static inline int pad32h(int x) { return (x + 31) & ~31; }

extern "C" void kernel_launch(void* const* d_in, const int* in_sizes, int n_in,
                              void* d_out, int out_size, void* d_ws, size_t ws_size,
                              hipStream_t stream)
{
  (void)in_sizes; (void)n_in; (void)out_size; (void)ws_size;

  // ---- model config (mirrors SA_CFG / FP_CFG) ----
  static const int   SA_np[4]        = {1024, 256, 64, 16};
  static const float SA_r[4][2]      = {{0.05f, 0.1f}, {0.1f, 0.2f}, {0.2f, 0.4f}, {0.4f, 0.8f}};
  static const int   SA_k[2]         = {16, 32};
  static const int   SA_cin[4]       = {9, 96, 256, 512};
  static const int   SA_mlp[4][2][3] = {{{16,16,32},{32,32,64}},
                                        {{64,64,128},{64,96,128}},
                                        {{128,196,256},{128,196,256}},
                                        {{256,256,512},{256,384,512}}};
  static const int   FP_cin[4]       = {1536, 512, 352, 128};      // fp4, fp3, fp2, fp1
  static const int   FP_nl[4]        = {2, 2, 2, 3};
  static const int   FP_mlp[4][3]    = {{256,256,0},{256,256,0},{256,128,0},{128,128,128}};

  // ---- workspace bump allocator ----
  char* wsp = (char*)d_ws;
  auto alloc = [&](size_t bytes) -> void* {
    void* r = (void*)wsp;
    wsp += (bytes + 255) & ~(size_t)255;
    return r;
  };

  // ---- unpack params (setup_inputs() dict order) + pre-pack per layer ----
  int p = 0;
  const float* xin = (const float*)d_in[p++];

  auto pack_layer = [&](Mlp& m, int L, int cin_l) {
    const float* W  = (const float*)d_in[p++];
    const float* bi = (const float*)d_in[p++];
    const float* ga = (const float*)d_in[p++];
    const float* be = (const float*)d_in[p++];
    const float* mn = (const float*)d_in[p++];
    const float* vr = (const float*)d_in[p++];
    const int cout = m.cout[L];
    const int nK = (cin_l + 31) >> 5;
    const int coutPad = pad32h(cout);
    const int nT = coutPad >> 4;
    const int total = nT * nK * 512;
    _Float16* pW = (_Float16*)alloc((size_t)total * sizeof(_Float16));
    float* es = (float*)alloc((size_t)coutPad * sizeof(float));
    float* eb = (float*)alloc((size_t)coutPad * sizeof(float));
    pack_w_kernel<<<(total + 255) / 256, 256, 0, stream>>>(W, pW, cin_l, cout);
    pack_ep_kernel<<<(coutPad + 255) / 256, 256, 0, stream>>>(bi, ga, be, mn, vr, es, eb, cout, coutPad);
    m.pW[L] = pW; m.es[L] = es; m.eb[L] = eb;
  };

  Mlp sa[4][2];
  for (int i = 0; i < 4; ++i)
    for (int j = 0; j < 2; ++j) {
      sa[i][j].nlayers = 3;
      sa[i][j].cin = SA_cin[i] + 3;
      for (int L = 0; L < 3; ++L) sa[i][j].cout[L] = SA_mlp[i][j][L];
      int cin_l = sa[i][j].cin;
      for (int L = 0; L < 3; ++L) { pack_layer(sa[i][j], L, cin_l); cin_l = sa[i][j].cout[L]; }
    }
  Mlp fp[4];
  for (int i = 0; i < 4; ++i) {
    fp[i].nlayers = FP_nl[i];
    fp[i].cin = FP_cin[i];
    for (int L = 0; L < FP_nl[i]; ++L) fp[i].cout[L] = FP_mlp[i][L];
    int cin_l = fp[i].cin;
    for (int L = 0; L < FP_nl[i]; ++L) { pack_layer(fp[i], L, cin_l); cin_l = fp[i].cout[L]; }
  }
  Mlp head; head.nlayers = 1; head.cin = 128; head.cout[0] = 128;
  pack_layer(head, 0, 128);
  const float* conv2_W = (const float*)d_in[p++];
  const float* conv2_b = (const float*)d_in[p++];

  // ---- output layout: kp | softmax^T | l4_p ----
  float* out_f = (float*)d_out;
  float* kp    = out_f;
  float* xT    = out_f + (size_t)BATCH * NCLS * 3;
  float* l4_p  = xT + (size_t)BATCH * NCLS * NPTS;     // (B,16,1024)

  // ---- activations workspace ----
  float* l0_xyz = (float*)alloc((size_t)BATCH * NPTS * 3 * 4);
  float* l0_pts = (float*)alloc((size_t)BATCH * NPTS * 9 * 4);
  float* dist   = (float*)alloc((size_t)BATCH * NPTS * 4);
  int*   fidx   = (int*)  alloc((size_t)BATCH * 1024 * 4);
  int*   gidx   = (int*)  alloc((size_t)BATCH * 1024 * 32 * 4);
  float* l1_xyz = (float*)alloc((size_t)BATCH * 1024 * 3 * 4);
  float* l2_xyz = (float*)alloc((size_t)BATCH * 256 * 3 * 4);
  float* l3_xyz = (float*)alloc((size_t)BATCH * 64 * 3 * 4);
  float* l4_xyz = (float*)alloc((size_t)BATCH * 16 * 3 * 4);
  float* l1_p   = (float*)alloc((size_t)BATCH * 1024 * 96 * 4);
  float* l2_p   = (float*)alloc((size_t)BATCH * 256 * 256 * 4);
  float* l3_p   = (float*)alloc((size_t)BATCH * 64 * 512 * 4);
  float* l3_p2  = (float*)alloc((size_t)BATCH * 64 * 256 * 4);
  float* l2_p2  = (float*)alloc((size_t)BATCH * 256 * 256 * 4);
  float* l1_p2  = (float*)alloc((size_t)BATCH * 1024 * 128 * 4);
  float* l0_p   = (float*)alloc((size_t)BATCH * NPTS * 128 * 4);
  float* interp = (float*)alloc((size_t)BATCH * NPTS * 128 * 4); // max FP interp need
  float* logits = (float*)alloc((size_t)BATCH * NPTS * NCLS * 4);
  float* hbuf   = interp;  // head output reuses interp (interp dead by then)

  // ---- stage 0: transpose ----
  {
    int tot = BATCH * NPTS;
    transpose_in_kernel<<<(tot + 255) / 256, 256, 0, stream>>>(xin, l0_xyz, l0_pts);
  }

  // ---- SA stages ----
  const float* in_xyz[4] = {l0_xyz, l1_xyz, l2_xyz, l3_xyz};
  const float* in_pts[4] = {l0_pts, l1_p, l2_p, l3_p};
  float* st_xyz[4] = {l1_xyz, l2_xyz, l3_xyz, l4_xyz};
  float* st_out[4] = {l1_p, l2_p, l3_p, l4_p};
  const int st_n[4]       = {NPTS, 1024, 256, 64};
  const int st_ostr[4]    = {96, 256, 512, 1024};
  const int st_ooff[4][2] = {{0, 32}, {0, 128}, {0, 256}, {0, 512}};

  for (int i = 0; i < 4; ++i) {
    const int n = st_n[i], np = SA_np[i];
    fps_kernel<<<BATCH, 256, 0, stream>>>(in_xyz[i], fidx, dist, n, np);
    {
      int tot = BATCH * np;
      gather3_kernel<<<(tot + 255) / 256, 256, 0, stream>>>(in_xyz[i], fidx, st_xyz[i], n, np);
    }
    for (int j = 0; j < 2; ++j) {
      const int k = SA_k[j];
      const float r2 = SA_r[i][j] * SA_r[i][j];
      int tot = BATCH * np;
      ball_query_kernel<<<(tot + 255) / 256, 256, 0, stream>>>(in_xyz[i], st_xyz[i], gidx, n, np, k, r2);
      int stride = pad32h(SA_cin[i] + 3);
      for (int L = 0; L < 3; ++L) stride = max(stride, pad32h(SA_mlp[i][j][L]));
      size_t lds = (size_t)2 * k * stride * sizeof(_Float16);
      sa_branch_kernel<<<BATCH * np, 128, lds, stream>>>(
          in_xyz[i], in_pts[i], st_xyz[i], gidx, sa[i][j], st_out[i],
          n, np, k, SA_cin[i], stride, st_ostr[i], st_ooff[i][j]);
    }
  }

  // ---- FP stages ----
  struct FpRun { const float* x1; int n1; const float* x2; int s;
                 const float* p1; int c1; const float* p2; int c2; float* out; };
  FpRun runs[4] = {
      {l3_xyz, 64,   l4_xyz, 16,  l3_p, 512, l4_p,  1024, l3_p2},
      {l2_xyz, 256,  l3_xyz, 64,  l2_p, 256, l3_p2, 256,  l2_p2},
      {l1_xyz, 1024, l2_xyz, 256, l1_p, 96,  l2_p2, 256,  l1_p2},
      {l0_xyz, NPTS, l1_xyz, 1024, nullptr, 0, l1_p2, 128, l0_p},
  };
  for (int i = 0; i < 4; ++i) {
    FpRun& r = runs[i];
    interp3_kernel<<<BATCH * r.n1, 32, 0, stream>>>(r.x1, r.x2, r.p2, interp, r.n1, r.s, r.c2);
    int stride = pad32h(r.c1 + r.c2);
    for (int L = 0; L < FP_nl[i]; ++L) stride = max(stride, pad32h(FP_mlp[i][L]));
    size_t lds = (size_t)2 * 16 * stride * sizeof(_Float16);
    fp_mlp_kernel<<<(BATCH * r.n1) / 16, 128, lds, stream>>>(
        r.p1, r.c1, interp, r.c2, fp[i], r.out, stride);
  }

  // ---- head + conv2 + softmax/kp ----
  {
    int stride = 128;
    size_t lds = (size_t)2 * 16 * stride * sizeof(_Float16);
    fp_mlp_kernel<<<(BATCH * NPTS) / 16, 128, lds, stream>>>(
        l0_p, 128, nullptr, 0, head, hbuf, stride);
  }
  {
    int tot = BATCH * NPTS;
    conv2_kernel<<<(tot + 255) / 256, 256, 0, stream>>>(hbuf, conv2_W, conv2_b, logits);
  }
  softmax_kp_kernel<<<BATCH * NCLS, 256, 0, stream>>>(logits, l0_xyz, xT, kp, NPTS);
}
